// Linear_13340168422051
// MI455X (gfx1250) — 8-bit matrix operand order repaired and checked by reading, compiled, not run on hardware
//
#include <hip/hip_runtime.h>
#include <hip/hip_bf16.h>
#include <stdint.h>

typedef __attribute__((ext_vector_type(8))) int v8i;

#define TILE_M 128
#define TILE_N 128
#define KSTEP  64
#define LROW   80
#define NBUF   4

__device__ __forceinline__ unsigned lds_off(const void* p) {
    return (unsigned)(unsigned long long)p;
}

__device__ __forceinline__ void async_load_x4(unsigned l0, unsigned l1,
                                              unsigned l2, unsigned l3,
                                              const void* g0, const void* g1,
                                              const void* g2, const void* g3) {
    asm volatile(
        "s_clause 0x3\n\t"
        "global_load_async_to_lds_b128 %0, %4, off\n\t"
        "global_load_async_to_lds_b128 %1, %5, off\n\t"
        "global_load_async_to_lds_b128 %2, %6, off\n\t"
        "global_load_async_to_lds_b128 %3, %7, off"
        :: "v"(l0), "v"(l1), "v"(l2), "v"(l3),
           "v"((unsigned long long)g0), "v"((unsigned long long)g1),
           "v"((unsigned long long)g2), "v"((unsigned long long)g3)
        : "memory");
}

__device__ __forceinline__ void wait_async0() { asm volatile("s_wait_asynccnt 0x0" ::: "memory"); }
__device__ __forceinline__ void wait_async4() { asm volatile("s_wait_asynccnt 0x4" ::: "memory"); }
__device__ __forceinline__ void wait_async8() { asm volatile("s_wait_asynccnt 0x8" ::: "memory"); }

__global__ __launch_bounds__(256)
void quantize_rows_kernel(const float* __restrict__ x,
                          int8_t* __restrict__ aq,
                          float* __restrict__ a_s,
                          int K) {
    __shared__ float red[256];
    const int row = blockIdx.x;
    const int t   = threadIdx.x;
    const float4* xr = (const float4*)(x + (size_t)row * K);

    float4 v[4];
    float m = 0.0f;
#pragma unroll
    for (int i = 0; i < 4; ++i) {
        v[i] = xr[t + i * 256];
        m = fmaxf(m, fmaxf(fmaxf(fabsf(v[i].x), fabsf(v[i].y)),
                           fmaxf(fabsf(v[i].z), fabsf(v[i].w))));
    }
    red[t] = m;
    __syncthreads();
#pragma unroll
    for (int s = 128; s > 0; s >>= 1) {
        if (t < s) red[t] = fmaxf(red[t], red[t + s]);
        __syncthreads();
    }
    const float mag = red[0];
    const float scl = (mag == 0.0f) ? 1.0f : mag * (1.0f / 127.0f);
    const float inv = 1.0f / scl;
    if (t == 0) a_s[row] = scl;

    int* aqi = (int*)(aq + (size_t)row * K);
#pragma unroll
    for (int i = 0; i < 4; ++i) {
        float q0 = fminf(127.0f, fmaxf(-128.0f, rintf(v[i].x * inv)));
        float q1 = fminf(127.0f, fmaxf(-128.0f, rintf(v[i].y * inv)));
        float q2 = fminf(127.0f, fmaxf(-128.0f, rintf(v[i].z * inv)));
        float q3 = fminf(127.0f, fmaxf(-128.0f, rintf(v[i].w * inv)));
        int b0 = (int)q0, b1 = (int)q1, b2 = (int)q2, b3 = (int)q3;
        aqi[t + i * 256] = (b0 & 0xff) | ((b1 & 0xff) << 8) |
                           ((b2 & 0xff) << 16) | ((b3 & 0xff) << 24);
    }
}

__global__ __launch_bounds__(256)
void gemm_i8_wmma_kernel(const int8_t* __restrict__ Aq,
                         const int8_t* __restrict__ W,
                         const float* __restrict__ a_s,
                         const float* __restrict__ wscale,
                         float* __restrict__ out,
                         int M, int N, int K) {
    __shared__ int8_t As[NBUF][TILE_M * LROW];
    __shared__ int8_t Bs[NBUF][TILE_N * LROW];

    const int tid    = threadIdx.x;
    const int wave   = tid >> 5;
    const int lane   = tid & 31;
    const int half   = lane >> 4;
    const int r      = lane & 15;
    const int wave_m = wave >> 1;
    const int wave_n = wave & 1;

    const int m_blk = blockIdx.y * TILE_M;
    const int n_blk = blockIdx.x * TILE_N;

    const int srow = tid >> 2;
    const int sseg = (tid & 3) * 16;
    const int8_t* gA0 = Aq + (size_t)(m_blk + srow)      * K + sseg;
    const int8_t* gA1 = Aq + (size_t)(m_blk + srow + 64) * K + sseg;
    const int8_t* gB0 = W  + (size_t)(n_blk + srow)      * K + sseg;
    const int8_t* gB1 = W  + (size_t)(n_blk + srow + 64) * K + sseg;
    unsigned lA[NBUF], lB[NBUF];
#pragma unroll
    for (int b = 0; b < NBUF; ++b) {
        lA[b] = lds_off(&As[b][srow * LROW + sseg]);
        lB[b] = lds_off(&Bs[b][srow * LROW + sseg]);
    }

    const v8i vzero = {0, 0, 0, 0, 0, 0, 0, 0};
    v8i acc[2][4];
#pragma unroll
    for (int mt = 0; mt < 2; ++mt)
#pragma unroll
        for (int nt = 0; nt < 4; ++nt)
            acc[mt][nt] = vzero;

    const int8_t* abase[NBUF];
    const int8_t* bbase[NBUF];
#pragma unroll
    for (int b = 0; b < NBUF; ++b) {
        abase[b] = &As[b][(wave_m * 32 + r) * LROW + half * 16];
        bbase[b] = &Bs[b][(wave_n * 64 + r) * LROW + half * 16];
    }

    auto stage = [&](unsigned laA, unsigned laB, int k0) {
        async_load_x4(laA, laA + 64 * LROW, laB, laB + 64 * LROW,
                      gA0 + k0, gA1 + k0, gB0 + k0, gB1 + k0);
    };

    auto compute_step = [&](const int8_t* ab, const int8_t* bb) {
        v8i afrag[2];
#pragma unroll
        for (int mt = 0; mt < 2; ++mt) {
            const int8_t* ap = ab + mt * 16 * LROW;
            int4 a03 = *(const int4*)(ap + 0);
            int4 a47 = *(const int4*)(ap + 32);
            v8i a = {a03.x, a03.y, a03.z, a03.w, a47.x, a47.y, a47.z, a47.w};
            afrag[mt] = a;
        }
        v8i bfrag[4];
#pragma unroll
        for (int nt = 0; nt < 4; ++nt) {
            const int8_t* bp = bb + nt * 16 * LROW;
            int4 b03 = *(const int4*)(bp + 0);
            int4 b47 = *(const int4*)(bp + 32);
            v8i b = {b03.x, b03.y, b03.z, b03.w, b47.x, b47.y, b47.z, b47.w};
            bfrag[nt] = b;
        }
#pragma unroll
        for (int nt = 0; nt < 4; ++nt)
#pragma unroll
            for (int mt = 0; mt < 2; ++mt) {
                acc[mt][nt] = __builtin_amdgcn_wmma_i32_16x16x64_iu8(
                    true, afrag[mt], true, bfrag[nt], acc[mt][nt], false, false);
            }
    };

    const int NK = K / KSTEP;

    stage(lA[0], lB[0], 0);
    stage(lA[1], lB[1], KSTEP);

    for (int kb = 0; kb < NK - 4; kb += 4) {
#pragma unroll
        for (int u = 0; u < 4; ++u) {
            const int sbuf = (u + 2) & 3;
            stage(lA[sbuf], lB[sbuf], (kb + u + 2) * KSTEP);
            wait_async8();
            __syncthreads();
            compute_step(abase[u], bbase[u]);
        }
    }

    stage(lA[2], lB[2], (NK - 2) * KSTEP);
    wait_async8();
    __syncthreads();
    compute_step(abase[0], bbase[0]);

    stage(lA[3], lB[3], (NK - 1) * KSTEP);
    wait_async8();
    __syncthreads();
    compute_step(abase[1], bbase[1]);

    wait_async4();
    __syncthreads();
    compute_step(abase[2], bbase[2]);

    wait_async0();
    __syncthreads();
    compute_step(abase[3], bbase[3]);

    const int m_wave = m_blk + wave_m * 32;
    const int n_wave = n_blk + wave_n * 64;
#pragma unroll
    for (int mt = 0; mt < 2; ++mt) {
        const int row0 = m_wave + mt * 16 + half * 8;
        float asv[8];
#pragma unroll
        for (int j = 0; j < 8; ++j) asv[j] = a_s[row0 + j];
#pragma unroll
        for (int nt = 0; nt < 4; ++nt) {
            const int col = n_wave + nt * 16 + r;
            const float ws = wscale[col];
            float* op = out + (size_t)row0 * N + col;
#pragma unroll
            for (int j = 0; j < 8; ++j)
                op[(size_t)j * N] = (float)acc[mt][nt][j] * asv[j] * ws;
        }
    }
}

extern "C" void kernel_launch(void* const* d_in, const int* in_sizes, int n_in,
                              void* d_out, int out_size, void* d_ws, size_t ws_size,
                              hipStream_t stream) {
    const float*  x      = (const float*)d_in[0];
    const int8_t* weight = (const int8_t*)d_in[1];
    const float*  wscale = (const float*)d_in[2];
    float*        out    = (float*)d_out;

    const int N = in_sizes[2];
    const int K = in_sizes[1] / N;
    const int M = (int)((long long)in_sizes[0] / K);

    int8_t* Aq  = (int8_t*)d_ws;
    float*  a_s = (float*)((char*)d_ws + (size_t)M * K);

    quantize_rows_kernel<<<M, 256, 0, stream>>>(x, Aq, a_s, K);

    dim3 grid(N / TILE_N, M / TILE_M);
    gemm_i8_wmma_kernel<<<grid, 256, 0, stream>>>(Aq, weight, a_s, wscale, out,
                                                  M, N, K);
}
